// Attention_4037269258988
// MI455X (gfx1250) — matrix operand order repaired and checked by reading, not compiled, not run on hardware
//
#include <hip/hip_runtime.h>
#include <hip/hip_bf16.h>

typedef __attribute__((ext_vector_type(16))) _Float16 v16h;
typedef __attribute__((ext_vector_type(8)))  float    v8f;
typedef _Float16 half_t;

constexpr int cB = 2, cN = 2048, cDIM = 512, cH = 8, cDH = 64, cINNER = 512;
constexpr int NX  = cB * cN * cDIM;        // 2,097,152  x -> f16
constexpr int NW1 = cDIM * 3 * cINNER;     //   786,432  W_qkv^T f16
constexpr int NW2 = cINNER * cDIM;         //   262,144  W_out^T f16
constexpr int NH  = cB * cH * cN * cDH;    // 2,097,152  each of Q,K,Vt
constexpr int NO  = cB * cN * cINNER;      // 2,097,152  attn output f16

union Frag16 { v16h h; unsigned int u[8]; };

__device__ inline v8f vzero8() {
    v8f z;
#pragma unroll
    for (int r = 0; r < 8; ++r) z[r] = 0.f;
    return z;
}

__device__ inline v8f wmma_f16(v16h a, v16h b, v8f c) {
    return __builtin_amdgcn_wmma_f32_16x16x32_f16(false, a, false, b, (short)0, c, false, false);
}

// 16-bit A fragment 16x32: lane holds row m = lane&15; VGPR v holds K pair
// k = (v>=4?16:0) + (lane>>4)*8 + (v&3)*2. Row-major A => pair is one dword,
// and the 8 dwords form two contiguous 16B runs -> 2x global_load_b128.
__device__ inline v16h load_a_frag(const half_t* __restrict__ A, int lda, int lane) {
    Frag16 f;
    int m = lane & 15, hx = lane >> 4;
    const half_t* row = A + (long)m * lda;
#pragma unroll
    for (int v = 0; v < 8; ++v) {
        int k = ((v & 4) << 2) + hx * 8 + ((v & 3) << 1);
        f.u[v] = *(const unsigned int*)(row + k);
    }
    return f.h;
}

// 16-bit B fragment 32x16: lane holds col n = lane&15; VGPR v holds K pair
// k = (v>=4?16:0) + (lane>>4)*8 + (v&3)*2. Bt is stored [n][k] (K contiguous).
__device__ inline v16h load_b_frag(const half_t* __restrict__ Bt, int ldb, int lane) {
    Frag16 f;
    int n = lane & 15, hx = lane >> 4;
    const half_t* row = Bt + (long)n * ldb + hx * 8;
#pragma unroll
    for (int v = 0; v < 8; ++v)
        f.u[v] = *(const unsigned int*)(row + ((v & 4) << 2) + ((v & 3) << 1));
    return f.h;
}

// ---------------- prep: f16 conversion + weight transposes ----------------
__global__ void __launch_bounds__(256) prep_kernel(
    const float* __restrict__ x, const float* __restrict__ Wqkv,
    const float* __restrict__ Wout, half_t* __restrict__ xh,
    half_t* __restrict__ Wqkvt, half_t* __restrict__ Woutt) {
    int idx = blockIdx.x * blockDim.x + threadIdx.x;
    int stride = gridDim.x * blockDim.x;
    int total = NX + NW1 + NW2;
    for (; idx < total; idx += stride) {
        if (idx < NX) {
            xh[idx] = (half_t)x[idx];
        } else if (idx < NX + NW1) {
            int t = idx - NX;              // t = n*512 + k
            int n = t >> 9, k = t & 511;
            Wqkvt[t] = (half_t)Wqkv[(long)k * (3 * cINNER) + n];
        } else {
            int t = idx - NX - NW1;
            int n = t >> 9, k = t & 511;
            Woutt[t] = (half_t)Wout[(long)k * cDIM + n];
        }
    }
}

// ---------------- QKV GEMM: (B*N x 512) x (512 x 1536) ----------------
// wave computes 16 rows x 64 cols; manual 2x ping-pong unroll so fragment
// registers are written directly by loads (no cur<-nxt copy chains).
__global__ void __launch_bounds__(256) qkv_kernel(
    const half_t* __restrict__ xh, const half_t* __restrict__ Wt,
    half_t* __restrict__ Qh, half_t* __restrict__ Kh, half_t* __restrict__ Vt) {
    int wave = (blockIdx.x * blockDim.x + threadIdx.x) >> 5;
    int lane = threadIdx.x & 31;
    int strip = wave % 24;            // 24 strips of 64 cols over 1536
    int rt = wave / 24;               // 256 row tiles of 16 over 4096
    int row0 = rt * 16, col0 = strip * 64;

    v8f acc[4];
#pragma unroll
    for (int c = 0; c < 4; ++c) acc[c] = vzero8();

    const half_t* Abase = xh + (long)row0 * cDIM;
    const half_t* Bbase = Wt + (long)col0 * cDIM;

    v16h aA = load_a_frag(Abase, cDIM, lane);
    v16h bA[4];
#pragma unroll
    for (int c = 0; c < 4; ++c)
        bA[c] = load_b_frag(Bbase + (long)c * 16 * cDIM, cDIM, lane);

    for (int kk = 0; kk < cDIM; kk += 64) {
        int k1 = kk + 32;
        v16h aB = load_a_frag(Abase + k1, cDIM, lane);
        v16h bB[4];
#pragma unroll
        for (int c = 0; c < 4; ++c)
            bB[c] = load_b_frag(Bbase + (long)c * 16 * cDIM + k1, cDIM, lane);
#pragma unroll
        for (int c = 0; c < 4; ++c)
            acc[c] = wmma_f16(aA, bA[c], acc[c]);

        int k2 = (kk + 64 < cDIM) ? kk + 64 : kk;   // clamped prefetch
        aA = load_a_frag(Abase + k2, cDIM, lane);
#pragma unroll
        for (int c = 0; c < 4; ++c)
            bA[c] = load_b_frag(Bbase + (long)c * 16 * cDIM + k2, cDIM, lane);
#pragma unroll
        for (int c = 0; c < 4; ++c)
            acc[c] = wmma_f16(aB, bB[c], acc[c]);
    }

    int hx = lane >> 4, nl = lane & 15;
#pragma unroll
    for (int c = 0; c < 4; ++c) {
        int col = col0 + c * 16 + nl;
        int sect = col >> 9;          // 0=Q 1=K 2=V
        int cc = col & 511;
        int h = cc >> 6, d = cc & 63;
#pragma unroll
        for (int r = 0; r < 8; ++r) {
            int row = row0 + r + hx * 8;
            int b = row >> 11, i = row & 2047;
            long bh = (long)(b * cH + h);
            half_t val = (half_t)acc[c][r];
            if (sect == 0)      Qh[(bh * cN + i) * cDH + d] = val;
            else if (sect == 1) Kh[(bh * cN + i) * cDH + d] = val;
            else                Vt[(bh * cDH + d) * cN + i] = val;
        }
    }
}

// ---------------- fused flash attention ----------------
// one wave per (b,h,16-row tile); computes S^T = K·Q^T so exp(S^T) packs
// lane-locally into the A fragment of the P×V WMMA (no transpose needed).
// j-tiles of 32 processed in ping-pong pairs: each tile's K/V fragments and
// mask word are loaded one stage ahead, hidden under WMMA + softmax work.
struct KVTile {
    v16h ka00, ka01, ka10, ka11;
    v16h vb0, vb1, vb2, vb3;
    bool aj;
};

__device__ inline KVTile load_tile(const half_t* __restrict__ Kbase,
                                   const half_t* __restrict__ Vbase,
                                   const int* __restrict__ np_row,
                                   const int* __restrict__ bert_row,
                                   int jb, int lane) {
    KVTile t;
    int jl = jb + lane;
    t.aj   = (np_row[jl] != 0) && (bert_row[jl] != 1);
    t.ka00 = load_a_frag(Kbase + (long)jb * cDH, cDH, lane);
    t.ka01 = load_a_frag(Kbase + (long)jb * cDH + 32, cDH, lane);
    t.ka10 = load_a_frag(Kbase + (long)(jb + 16) * cDH, cDH, lane);
    t.ka11 = load_a_frag(Kbase + (long)(jb + 16) * cDH + 32, cDH, lane);
    t.vb0  = load_b_frag(Vbase + (long)0  * cN + jb, cN, lane);
    t.vb1  = load_b_frag(Vbase + (long)16 * cN + jb, cN, lane);
    t.vb2  = load_b_frag(Vbase + (long)32 * cN + jb, cN, lane);
    t.vb3  = load_b_frag(Vbase + (long)48 * cN + jb, cN, lane);
    return t;
}

__device__ inline void process_tile(const KVTile& t, v16h qb0, v16h qb1,
                                    bool row_ok, int hx, int lane,
                                    float& mrow, float& lrow, v8f (&o)[4]) {
    const float scale = 0.125f;       // 1/sqrt(64)
    unsigned int jm = (t.aj ? 1u : 0u) << lane;      // wave-wide column mask
#pragma unroll
    for (int s = 16; s > 0; s >>= 1) jm |= __shfl_xor(jm, s, 32);

    v8f s0 = vzero8(), s1 = vzero8();
    s0 = wmma_f16(t.ka00, qb0, s0);
    s0 = wmma_f16(t.ka01, qb1, s0);
    s1 = wmma_f16(t.ka10, qb0, s1);
    s1 = wmma_f16(t.ka11, qb1, s1);

    float p0[8], p1[8], rmax = -1e30f;
#pragma unroll
    for (int r = 0; r < 8; ++r) {
        int j0 = r + hx * 8;          // S^T row (= key index) local, tile0
        int j1 = j0 + 16;
        float v0 = (row_ok && ((jm >> j0) & 1u)) ? s0[r] * scale : -1000.f;
        float v1 = (row_ok && ((jm >> j1) & 1u)) ? s1[r] * scale : -1000.f;
        p0[r] = v0; p1[r] = v1;
        rmax = fmaxf(rmax, fmaxf(v0, v1));
    }
    rmax = fmaxf(rmax, __shfl_xor(rmax, 16, 32));
    float mnew = fmaxf(mrow, rmax);
    float corr = __expf(mrow - mnew);
    float rsum = 0.f;
#pragma unroll
    for (int r = 0; r < 8; ++r) {
        p0[r] = __expf(p0[r] - mnew);
        p1[r] = __expf(p1[r] - mnew);
        rsum += p0[r] + p1[r];
    }
    rsum += __shfl_xor(rsum, 16, 32);
    lrow = lrow * corr + rsum;
    mrow = mnew;

#pragma unroll
    for (int r = 0; r < 8; ++r) {              // rescale O rows (row = r+8*hx)
        float cr = __shfl(corr, r + hx * 8, 32);
#pragma unroll
        for (int dt = 0; dt < 4; ++dt) o[dt][r] *= cr;
    }

    Frag16 pa;                                  // lane-local P pack: A 16x32
#pragma unroll
    for (int r = 0; r < 8; ++r) {
        pa.h[r]     = (_Float16)p0[r];
        pa.h[8 + r] = (_Float16)p1[r];
    }
    o[0] = wmma_f16(pa.h, t.vb0, o[0]);
    o[1] = wmma_f16(pa.h, t.vb1, o[1]);
    o[2] = wmma_f16(pa.h, t.vb2, o[2]);
    o[3] = wmma_f16(pa.h, t.vb3, o[3]);
}

__global__ void __launch_bounds__(256) attn_kernel(
    const half_t* __restrict__ Qh, const half_t* __restrict__ Kh,
    const half_t* __restrict__ Vt, const int* __restrict__ mask_np,
    const int* __restrict__ mask_bert, half_t* __restrict__ Oh) {
    int wave = (blockIdx.x * blockDim.x + threadIdx.x) >> 5;
    int lane = threadIdx.x & 31;
    int it = wave & 127;              // 128 row tiles
    int bh = wave >> 7;               // b*8+h
    int b = bh >> 3, h = bh & 7;
    int ib = it * 16;
    int hx = lane >> 4, il = lane & 15;

    const half_t* Qbase = Qh + ((long)bh * cN + ib) * cDH;
    const half_t* Kbase = Kh + (long)bh * cN * cDH;
    const half_t* Vbase = Vt + (long)bh * cDH * cN;
    const int* np_row   = mask_np + b * cN;
    const int* bert_row = mask_bert + b * cN;

    bool row_ok = np_row[ib + il] != 0;

    v16h qb0 = load_b_frag(Qbase, cDH, lane);        // d = 0..31
    v16h qb1 = load_b_frag(Qbase + 32, cDH, lane);   // d = 32..63

    float mrow = -1e30f, lrow = 0.f;
    v8f o[4];
#pragma unroll
    for (int dt = 0; dt < 4; ++dt) o[dt] = vzero8();

    KVTile tA = load_tile(Kbase, Vbase, np_row, bert_row, 0, lane);
    for (int jb = 0; jb < cN; jb += 64) {
        KVTile tB = load_tile(Kbase, Vbase, np_row, bert_row, jb + 32, lane);
        process_tile(tA, qb0, qb1, row_ok, hx, lane, mrow, lrow, o);

        int jn = (jb + 64 < cN) ? jb + 64 : jb;     // clamped prefetch
        tA = load_tile(Kbase, Vbase, np_row, bert_row, jn, lane);
        process_tile(tB, qb0, qb1, row_ok, hx, lane, mrow, lrow, o);
    }

    float linv = 1.f / lrow;
#pragma unroll
    for (int r = 0; r < 8; ++r) {
        float lr = __shfl(linv, r + hx * 8, 32);
        int irow = ib + r + hx * 8;
#pragma unroll
        for (int dt = 0; dt < 4; ++dt) {
            Oh[((long)b * cN + irow) * cINNER + h * cDH + dt * 16 + il] =
                (half_t)(o[dt][r] * lr);
        }
    }
}

// ---------------- output GEMM: (B*N x 512) x (512 x 512) + bias ----------------
__global__ void __launch_bounds__(256) out_kernel(
    const half_t* __restrict__ Oh, const half_t* __restrict__ Wt,
    const float* __restrict__ bias, float* __restrict__ out) {
    int wave = (blockIdx.x * blockDim.x + threadIdx.x) >> 5;
    int lane = threadIdx.x & 31;
    int strip = wave & 7;             // 8 strips of 64 over 512
    int rt = wave >> 3;               // 256 row tiles
    int row0 = rt * 16, col0 = strip * 64;

    v8f acc[4];
#pragma unroll
    for (int c = 0; c < 4; ++c) acc[c] = vzero8();

    const half_t* Abase = Oh + (long)row0 * cINNER;
    const half_t* Bbase = Wt + (long)col0 * cINNER;

    v16h aA = load_a_frag(Abase, cINNER, lane);
    v16h bA[4];
#pragma unroll
    for (int c = 0; c < 4; ++c)
        bA[c] = load_b_frag(Bbase + (long)c * 16 * cINNER, cINNER, lane);

    for (int kk = 0; kk < cINNER; kk += 64) {
        int k1 = kk + 32;
        v16h aB = load_a_frag(Abase + k1, cINNER, lane);
        v16h bB[4];
#pragma unroll
        for (int c = 0; c < 4; ++c)
            bB[c] = load_b_frag(Bbase + (long)c * 16 * cINNER + k1, cINNER, lane);
#pragma unroll
        for (int c = 0; c < 4; ++c)
            acc[c] = wmma_f16(aA, bA[c], acc[c]);

        int k2 = (kk + 64 < cINNER) ? kk + 64 : kk;
        aA = load_a_frag(Abase + k2, cINNER, lane);
#pragma unroll
        for (int c = 0; c < 4; ++c)
            bA[c] = load_b_frag(Bbase + (long)c * 16 * cINNER + k2, cINNER, lane);
#pragma unroll
        for (int c = 0; c < 4; ++c)
            acc[c] = wmma_f16(aB, bB[c], acc[c]);
    }

    int hx = lane >> 4, nl = lane & 15;
#pragma unroll
    for (int c = 0; c < 4; ++c) {
        int col = col0 + c * 16 + nl;
        float bv = bias[col];
#pragma unroll
        for (int r = 0; r < 8; ++r) {
            int row = row0 + r + hx * 8;
            out[(long)row * cDIM + col] = acc[c][r] + bv;
        }
    }
}

extern "C" void kernel_launch(void* const* d_in, const int* in_sizes, int n_in,
                              void* d_out, int out_size, void* d_ws, size_t ws_size,
                              hipStream_t stream) {
    const float* x        = (const float*)d_in[0];
    const int*   mask_np  = (const int*)d_in[1];
    const int*   mask_bert= (const int*)d_in[2];
    const float* Wqkv     = (const float*)d_in[3];
    const float* Wout     = (const float*)d_in[4];
    const float* bias     = (const float*)d_in[5];
    float* out = (float*)d_out;

    half_t* ws = (half_t*)d_ws;
    half_t* xh    = ws;               ws += NX;
    half_t* Wqkvt = ws;               ws += NW1;
    half_t* Woutt = ws;               ws += NW2;
    half_t* Qh    = ws;               ws += NH;
    half_t* Kh    = ws;               ws += NH;
    half_t* Vt    = ws;               ws += NH;
    half_t* Oh    = ws;               ws += NO;

    prep_kernel<<<2048, 256, 0, stream>>>(x, Wqkv, Wout, xh, Wqkvt, Woutt);
    // 256 row tiles * 24 col strips = 6144 waves -> 768 blocks of 8 waves
    qkv_kernel<<<768, 256, 0, stream>>>(xh, Wqkvt, Qh, Kh, Vt);
    // B*H*(N/16) = 2048 waves -> 256 blocks
    attn_kernel<<<256, 256, 0, stream>>>(Qh, Kh, Vt, mask_np, mask_bert, Oh);
    // 256 row tiles * 8 strips = 2048 waves -> 256 blocks
    out_kernel<<<256, 256, 0, stream>>>(Oh, Woutt, bias, out);
}
